// GNNPolicy_63032940036535
// MI455X (gfx1250) — matrix operand order repaired and checked by reading, not compiled, not run on hardware
//
#include <hip/hip_runtime.h>
#include <hip/hip_bf16.h>

// ---------------------------------------------------------------------------
// Types for CDNA5 WMMA / TDM
// ---------------------------------------------------------------------------
typedef __attribute__((ext_vector_type(16))) __bf16 v16bf;
typedef __attribute__((ext_vector_type(8)))  float  v8f;
typedef unsigned int u32x4 __attribute__((ext_vector_type(4)));
typedef int          i32x8 __attribute__((ext_vector_type(8)));
typedef int          i32x4 __attribute__((ext_vector_type(4)));

union AFragU { v16bf v; unsigned int u[8]; };

#if defined(__has_builtin)
#if __has_builtin(__builtin_amdgcn_tensor_load_to_lds) && __has_builtin(__builtin_amdgcn_s_wait_tensorcnt)
#define USE_TDM 1
#endif
#endif
#ifndef USE_TDM
#define USE_TDM 0
#endif

// ---------------------------------------------------------------------------
// Problem constants
// ---------------------------------------------------------------------------
#define GN     50000
#define GE     800000
#define G_INC  128
#define G_D1   256      // HEADS*HID
#define G_HID  64
#define G_H1   4
#define G_DEG  16

// ---------------------------------------------------------------------------
// f32 -> bf16 (round to nearest even)
// ---------------------------------------------------------------------------
__device__ __forceinline__ unsigned short f2bf(float f) {
    unsigned int u = __float_as_uint(f);
    u = u + 0x7FFFu + ((u >> 16) & 1u);
    return (unsigned short)(u >> 16);
}

__global__ void f32_to_bf16_kernel(const float* __restrict__ in,
                                   unsigned short* __restrict__ out, long n) {
    long i = (long)blockIdx.x * blockDim.x + threadIdx.x;
    if (i < n) out[i] = f2bf(in[i]);
}

// weights: f32 [K,Nc] -> bf16 transposed [Nc,K]
__global__ void f32_to_bf16_T_kernel(const float* __restrict__ in,
                                     unsigned short* __restrict__ out, int K, int Nc) {
    long i = (long)blockIdx.x * blockDim.x + threadIdx.x;
    if (i >= (long)K * Nc) return;
    int k = (int)(i / Nc), n = (int)(i % Nc);
    out[(long)n * K + k] = f2bf(in[i]);
}

__global__ void fill_f32_kernel(float* __restrict__ p, float v, long n) {
    long i = (long)blockIdx.x * blockDim.x + threadIdx.x;
    if (i < n) p[i] = v;
}

__global__ void fill_u64_kernel(unsigned long long* __restrict__ p, long n) {
    long i = (long)blockIdx.x * blockDim.x + threadIdx.x;
    if (i < n) p[i] = 0ull;
}

// ---------------------------------------------------------------------------
// TDM: issue a 2-D tile DMA (rows x rowlen bf16 elements) global -> LDS.
// Descriptor per CDNA5 D# group0/group1 bit layout (CDNA5 ISA).
// ---------------------------------------------------------------------------
#if USE_TDM
__device__ __forceinline__ void tdm_load_tile_2d(const unsigned short* gsrc,
                                                 unsigned lds_byte_off,
                                                 int rows, int rowlen_elems,
                                                 int stride_elems) {
    unsigned long long ga = (unsigned long long)gsrc;
    u32x4 g0;
    g0[0] = 1u;                                     // count=1, user mode, no gather
    g0[1] = lds_byte_off;                           // lds_addr  (bits 63:32)
    g0[2] = (unsigned)(ga & 0xFFFFFFFFull);         // global_addr[31:0]
    g0[3] = (unsigned)((ga >> 32) & 0x01FFFFFFull)  // global_addr[56:32]
          | (2u << 30);                             // type=2 ("image")
    i32x8 g1;
    g1[0] = 0x00010000;                             // wg_mask=0, data_size=1 (2B)
    g1[1] = (int)(((unsigned)stride_elems & 0xFFFFu) << 16);          // tensor_dim0[15:0]
    g1[2] = (int)((((unsigned)stride_elems >> 16) & 0xFFFFu)          // tensor_dim0[31:16]
          | (((unsigned)rows & 0xFFFFu) << 16));                      // tensor_dim1[15:0]
    g1[3] = (int)((((unsigned)rows >> 16) & 0xFFFFu)                  // tensor_dim1[31:16]
          | (((unsigned)rowlen_elems & 0xFFFFu) << 16));              // tile_dim0
    g1[4] = (int)((unsigned)rows & 0xFFFFu);                          // tile_dim1 (tile_dim2=0)
    g1[5] = stride_elems;                                             // tensor_dim0_stride[31:0]
    g1[6] = 0;                                                        // stride[47:32] | dim1_stride lo
    g1[7] = 0;
    i32x4 z4 = {0, 0, 0, 0};
#if __clang_major__ >= 23
    i32x8 z8 = {0, 0, 0, 0, 0, 0, 0, 0};
    __builtin_amdgcn_tensor_load_to_lds(g0, g1, z4, z4, z8, 0);
#else
    __builtin_amdgcn_tensor_load_to_lds(g0, g1, z4, z4, 0);
#endif
}
#endif

// ---------------------------------------------------------------------------
// WMMA GEMM:  C[M,Nc] = A[M,K](bf16) @ W[K,Nc] + bias[Nc]   (f32 out)
// W supplied TRANSPOSED as Wt[Nc,K] bf16.
// Block = 256 threads = 8 waves in 4(M) x 2(N); each wave: 16x32 D (2 accums).
// Block tile 64(M) x 64(N); K stepped by 32; tiles staged by TDM.
// ---------------------------------------------------------------------------
__global__ __launch_bounds__(256)
void wmma_gemm_bias_kernel(const unsigned short* __restrict__ A,
                           const unsigned short* __restrict__ Wt,
                           const float* __restrict__ bias,
                           float* __restrict__ C,
                           int M, int K, int Nc) {
    __shared__ __align__(16) unsigned short lA[64][32];    // m x k   (4 KB)
    __shared__ __align__(16) unsigned short lBt[64][32];   // n x k   (4 KB)

    const int tid  = threadIdx.x;
    const int wid  = tid >> 5;
    const int lane = tid & 31;
    const int half = lane >> 4;
    const int mr   = lane & 15;
    const int wm   = wid >> 1;         // 0..3
    const int wn   = wid & 1;          // 0..1

    const int m0 = blockIdx.x * 64;
    const int n0 = blockIdx.y * 64;

#if USE_TDM
    const unsigned ldsA = (unsigned)(unsigned long long)(void*)&lA[0][0];
    const unsigned ldsB = (unsigned)(unsigned long long)(void*)&lBt[0][0];
#endif

    v8f acc0 = {};
    v8f acc1 = {};

    for (int k0 = 0; k0 < K; k0 += 32) {
#if USE_TDM
        if (wid == 0) {
            tdm_load_tile_2d(A  + (long)m0 * K + k0, ldsA, 64, 32, K);
            tdm_load_tile_2d(Wt + (long)n0 * K + k0, ldsB, 64, 32, K);
            __builtin_amdgcn_s_wait_tensorcnt(0);
        }
#else
        {   // fallback: cooperative 16B loads (A rows clamped at M edge)
            int row = tid >> 2;                  // 0..63
            int ck  = (tid & 3) * 8;             // 0,8,16,24
            int gm  = m0 + row; if (gm > M - 1) gm = M - 1;
            *(uint4*)&lA[row][ck]  = *(const uint4*)(A  + ((long)gm * K + k0 + ck));
            *(uint4*)&lBt[row][ck] = *(const uint4*)(Wt + ((long)(n0 + row) * K + k0 + ck));
        }
#endif
        __syncthreads();

        // ---- build fragments (contiguous K order, same for A and B) ----
        AFragU a, b0, b1;
#pragma unroll
        for (int v = 0; v < 8; ++v) {
            // 16-bit A 16x32: lane-half owns contiguous K half (matches B)
            int ka = half * 16 + v * 2;
            a.u[v] = *(const unsigned int*)&lA[wm * 16 + mr][ka];
            // B 32x16 (16-bit): lane-half owns contiguous K half
            int kb = half * 16 + v * 2;
            b0.u[v] = *(const unsigned int*)&lBt[wn * 32 + mr][kb];
            b1.u[v] = *(const unsigned int*)&lBt[wn * 32 + 16 + mr][kb];
        }

        acc0 = __builtin_amdgcn_wmma_f32_16x16x32_bf16(
            false, a.v, false, b0.v, (short)0, acc0, false, false);
        acc1 = __builtin_amdgcn_wmma_f32_16x16x32_bf16(
            false, a.v, false, b1.v, (short)0, acc1, false, false);

        __syncthreads();
    }

    // ---- store D (32-bit C/D layout: M = r + 8*half, N = lane&15) ----
    const int nA = n0 + wn * 32 + mr;
    const int nB = nA + 16;
    const float bA = bias[nA];
    const float bB = bias[nB];
#pragma unroll
    for (int r = 0; r < 8; ++r) {
        int m = m0 + wm * 16 + half * 8 + r;
        if (m < M) {
            C[(long)m * Nc + nA] = acc0[r] + bA;
            C[(long)m * Nc + nB] = acc1[r] + bB;
        }
    }
}

// ---------------------------------------------------------------------------
// Edge attention scores:  e[E,H] = sum_c( leaky_relu(xl[src]+xr[dst]) * att )
// One wave per edge; lane covers channels/32 contiguous channels.
// ---------------------------------------------------------------------------
__global__ __launch_bounds__(256)
void edge_scores_kernel(const float* __restrict__ xl, const float* __restrict__ xr,
                        const int* __restrict__ src, const int* __restrict__ dst,
                        const float* __restrict__ att, float* __restrict__ e_out,
                        int E, int channels, int heads) {
    long gid  = (long)blockIdx.x * blockDim.x + threadIdx.x;
    int  edge = (int)(gid >> 5);
    int  lane = (int)(gid & 31);
    if (edge >= E) return;

    const int chunk = channels >> 5;           // channels per lane
    const int s = src[edge], d = dst[edge];
    const float* pl = xl + (long)s * channels + lane * chunk;
    const float* pr = xr + (long)d * channels + lane * chunk;
    const float* pa = att + lane * chunk;

    float sum = 0.f;
    for (int j = 0; j < chunk; ++j) {
        float h = pl[j] + pr[j];
        h = (h > 0.f) ? h : 0.2f * h;          // leaky_relu(0.2)
        sum += h * pa[j];
    }
    const int lph = 32 / heads;                // lanes per head
    for (int off = 1; off < lph; off <<= 1)
        sum += __shfl_xor(sum, off, 32);
    if ((lane & (lph - 1)) == 0)
        e_out[(long)edge * heads + lane / lph] = sum;
}

// ---------------------------------------------------------------------------
// Segment max (float atomic-max via int/uint trick)
// ---------------------------------------------------------------------------
__device__ __forceinline__ void atomicMaxF(float* addr, float val) {
    if (val >= 0.f) atomicMax((int*)addr, __float_as_int(val));
    else            atomicMin((unsigned int*)addr, __float_as_uint(val));
}

__global__ void seg_max_kernel(const float* __restrict__ e, const int* __restrict__ dst,
                               float* __restrict__ m, long EH, int heads) {
    long i = (long)blockIdx.x * blockDim.x + threadIdx.x;
    if (i >= EH) return;
    int edge = (int)(i / heads), h = (int)(i % heads);
    atomicMaxF(&m[(long)dst[edge] * heads + h], e[i]);
}

__global__ void seg_exp_sum_kernel(float* __restrict__ e, const int* __restrict__ dst,
                                   const float* __restrict__ m, float* __restrict__ ssum,
                                   long EH, int heads) {
    long i = (long)blockIdx.x * blockDim.x + threadIdx.x;
    if (i >= EH) return;
    int edge = (int)(i / heads), h = (int)(i % heads);
    long mi = (long)dst[edge] * heads + h;
    float a = __expf(e[i] - m[mi]);
    e[i] = a;                                   // overwrite score with exp value
    atomicAdd(&ssum[mi], a);
}

// ---------------------------------------------------------------------------
// Scatter:  out[dst] += alpha * xl[src]     (one wave per edge)
// ---------------------------------------------------------------------------
__global__ __launch_bounds__(256)
void edge_scatter_kernel(const float* __restrict__ a, const float* __restrict__ ssum,
                         const float* __restrict__ xl,
                         const int* __restrict__ src, const int* __restrict__ dst,
                         float* __restrict__ out, int E, int channels, int heads) {
    long gid  = (long)blockIdx.x * blockDim.x + threadIdx.x;
    int  edge = (int)(gid >> 5);
    int  lane = (int)(gid & 31);
    if (edge >= E) return;

    const int chunk = channels >> 5;
    const int lph   = 32 / heads;
    const int s = src[edge], d = dst[edge];
    const int h = lane / lph;
    float alpha = a[(long)edge * heads + h] /
                  (ssum[(long)d * heads + h] + 1e-16f);
    const float* pl = xl + (long)s * channels + lane * chunk;
    float*       po = out + (long)d * channels + lane * chunk;
    for (int j = 0; j < chunk; ++j)
        atomicAdd(&po[j], alpha * pl[j]);
}

// ---------------------------------------------------------------------------
// bias + ELU -> bf16 (layer 1 output feeding layer-2 WMMA GEMMs)
// ---------------------------------------------------------------------------
__global__ void bias_elu_bf16_kernel(const float* __restrict__ in, const float* __restrict__ bias,
                                     unsigned short* __restrict__ outbf, long n, int channels) {
    long i = (long)blockIdx.x * blockDim.x + threadIdx.x;
    if (i >= n) return;
    float v = in[i] + bias[i % channels];
    v = (v > 0.f) ? v : expm1f(v);
    outbf[i] = f2bf(v);
}

__global__ void bias_add_kernel(float* __restrict__ io, const float* __restrict__ bias,
                                long n, int channels) {
    long i = (long)blockIdx.x * blockDim.x + threadIdx.x;
    if (i < n) io[i] += bias[i % channels];
}

// ---------------------------------------------------------------------------
// argmax of x[:,3] -> packed (monotonic_key << 32) | ~idx  via atomicMax u64
// ---------------------------------------------------------------------------
__global__ void argmax_col_kernel(const float* __restrict__ x, unsigned long long* __restrict__ best,
                                  int Nn, int stride, int col) {
    long i = (long)blockIdx.x * blockDim.x + threadIdx.x;
    if (i >= Nn) return;
    unsigned int u = __float_as_uint(x[i * (long)stride + col]);
    unsigned int key = (u & 0x80000000u) ? ~u : (u | 0x80000000u);
    unsigned long long packed =
        ((unsigned long long)key << 32) | (unsigned int)(~(unsigned int)i);
    atomicMax(best, packed);
}

__global__ void score_neighbors_kernel(const unsigned long long* __restrict__ best,
                                       const int* __restrict__ edge_dst,
                                       const float* __restrict__ h2,
                                       const float* __restrict__ Ws,
                                       const float* __restrict__ bs,
                                       float* __restrict__ out, int deg, int hid) {
    int t = threadIdx.x;
    if (t >= deg) return;
    int cur = (int)(~(unsigned int)(*best & 0xFFFFFFFFull));
    int nb  = edge_dst[cur * deg + t];
    const float* p = h2 + (long)nb * hid;
    float s = 0.f;
    for (int k = 0; k < hid; ++k) s += p[k] * Ws[k];
    out[t] = s + bs[0];
}

// ---------------------------------------------------------------------------
// Launch
// ---------------------------------------------------------------------------
static inline int cdiv(long a, long b) { return (int)((a + b - 1) / b); }

extern "C" void kernel_launch(void* const* d_in, const int* in_sizes, int n_in,
                              void* d_out, int out_size, void* d_ws, size_t ws_size,
                              hipStream_t stream) {
    const float* x    = (const float*)d_in[0];
    const int*   esrc = (const int*)d_in[1];
    const int*   edst = (const int*)d_in[2];
    const float* Wl1  = (const float*)d_in[3];
    const float* bl1  = (const float*)d_in[4];
    const float* Wr1  = (const float*)d_in[5];
    const float* br1  = (const float*)d_in[6];
    const float* att1 = (const float*)d_in[7];
    const float* b1   = (const float*)d_in[8];
    const float* Wl2  = (const float*)d_in[9];
    const float* bl2  = (const float*)d_in[10];
    const float* Wr2  = (const float*)d_in[11];
    const float* br2  = (const float*)d_in[12];
    const float* att2 = (const float*)d_in[13];
    const float* b2   = (const float*)d_in[14];
    const float* Ws   = (const float*)d_in[15];
    const float* bs   = (const float*)d_in[16];
    float* out = (float*)d_out;

    // ---- workspace carving (lifetime-based reuse) ----
    char* ws = (char*)d_ws;
    size_t off = 0;
    auto alloc = [&](size_t bytes) -> char* {
        char* p = ws + off;
        off += (bytes + 255) & ~(size_t)255;
        return p;
    };
    unsigned short* xbf    = (unsigned short*)alloc((size_t)GN * G_INC * 2);  // 12.8 MB (reused as out2)
    unsigned short* wl1bfT = (unsigned short*)alloc((size_t)G_INC * G_D1 * 2);
    unsigned short* wr1bfT = (unsigned short*)alloc((size_t)G_INC * G_D1 * 2);
    unsigned short* wl2bfT = (unsigned short*)alloc((size_t)G_D1 * G_HID * 2);
    unsigned short* wr2bfT = (unsigned short*)alloc((size_t)G_D1 * G_HID * 2);
    float* xl1 = (float*)alloc((size_t)GN * G_D1 * 4);                        // 51.2 MB (reused as xl2|xr2)
    float* xr1 = (float*)alloc((size_t)GN * G_D1 * 4);                        // 51.2 MB (reused as h1bf)
    float* e1  = (float*)alloc((size_t)GE * G_H1 * 4);                        // 12.8 MB (reused as e2)
    float* m1  = (float*)alloc((size_t)GN * G_H1 * 4);
    float* s1  = (float*)alloc((size_t)GN * G_H1 * 4);
    float* out1 = (float*)alloc((size_t)GN * G_D1 * 4);                       // 51.2 MB
    unsigned long long* best = (unsigned long long*)alloc(8);
    (void)ws_size; (void)in_sizes; (void)n_in; (void)out_size;

    // lifetime reuse aliases
    unsigned short* h1bf = (unsigned short*)xr1;        // after layer-1 edge scores done
    float* xl2 = xl1;                                   // after layer-1 scatter done
    float* xr2 = xl1 + (size_t)GN * G_HID;
    float* e2  = e1;
    float* m2  = m1;
    float* s2  = s1;
    float* out2 = (float*)xbf;                          // after all GEMMs done

    const int TB = 256;

    // ---- bf16 conversions (weights pre-transposed for TDM-friendly tiles) ----
    f32_to_bf16_kernel<<<cdiv((long)GN * G_INC, TB), TB, 0, stream>>>(x, xbf, (long)GN * G_INC);
    f32_to_bf16_T_kernel<<<cdiv((long)G_INC * G_D1, TB), TB, 0, stream>>>(Wl1, wl1bfT, G_INC, G_D1);
    f32_to_bf16_T_kernel<<<cdiv((long)G_INC * G_D1, TB), TB, 0, stream>>>(Wr1, wr1bfT, G_INC, G_D1);
    f32_to_bf16_T_kernel<<<cdiv((long)G_D1 * G_HID, TB), TB, 0, stream>>>(Wl2, wl2bfT, G_D1, G_HID);
    f32_to_bf16_T_kernel<<<cdiv((long)G_D1 * G_HID, TB), TB, 0, stream>>>(Wr2, wr2bfT, G_D1, G_HID);

    // ---- layer 1: dense transforms (WMMA, TDM-staged tiles) ----
    {
        dim3 g(cdiv(GN, 64), G_D1 / 64);
        wmma_gemm_bias_kernel<<<g, TB, 0, stream>>>(xbf, wl1bfT, bl1, xl1, GN, G_INC, G_D1);
        wmma_gemm_bias_kernel<<<g, TB, 0, stream>>>(xbf, wr1bfT, br1, xr1, GN, G_INC, G_D1);
    }

    // ---- layer 1: edge attention + segment softmax + scatter ----
    edge_scores_kernel<<<cdiv((long)GE * 32, TB), TB, 0, stream>>>(xl1, xr1, esrc, edst, att1, e1, GE, G_D1, G_H1);
    fill_f32_kernel<<<cdiv((long)GN * G_H1, TB), TB, 0, stream>>>(m1, -3.0e38f, (long)GN * G_H1);
    fill_f32_kernel<<<cdiv((long)GN * G_H1, TB), TB, 0, stream>>>(s1, 0.f, (long)GN * G_H1);
    fill_f32_kernel<<<cdiv((long)GN * G_D1, TB), TB, 0, stream>>>(out1, 0.f, (long)GN * G_D1);
    seg_max_kernel<<<cdiv((long)GE * G_H1, TB), TB, 0, stream>>>(e1, edst, m1, (long)GE * G_H1, G_H1);
    seg_exp_sum_kernel<<<cdiv((long)GE * G_H1, TB), TB, 0, stream>>>(e1, edst, m1, s1, (long)GE * G_H1, G_H1);
    edge_scatter_kernel<<<cdiv((long)GE * 32, TB), TB, 0, stream>>>(e1, s1, xl1, esrc, edst, out1, GE, G_D1, G_H1);
    bias_elu_bf16_kernel<<<cdiv((long)GN * G_D1, TB), TB, 0, stream>>>(out1, b1, h1bf, (long)GN * G_D1, G_D1);

    // ---- layer 2: dense transforms (WMMA, TDM-staged tiles) ----
    {
        dim3 g(cdiv(GN, 64), G_HID / 64);
        wmma_gemm_bias_kernel<<<g, TB, 0, stream>>>(h1bf, wl2bfT, bl2, xl2, GN, G_D1, G_HID);
        wmma_gemm_bias_kernel<<<g, TB, 0, stream>>>(h1bf, wr2bfT, br2, xr2, GN, G_D1, G_HID);
    }

    // ---- layer 2: edge attention + segment softmax + scatter ----
    edge_scores_kernel<<<cdiv((long)GE * 32, TB), TB, 0, stream>>>(xl2, xr2, esrc, edst, att2, e2, GE, G_HID, 1);
    fill_f32_kernel<<<cdiv((long)GN, TB), TB, 0, stream>>>(m2, -3.0e38f, (long)GN);
    fill_f32_kernel<<<cdiv((long)GN, TB), TB, 0, stream>>>(s2, 0.f, (long)GN);
    fill_f32_kernel<<<cdiv((long)GN * G_HID, TB), TB, 0, stream>>>(out2, 0.f, (long)GN * G_HID);
    seg_max_kernel<<<cdiv((long)GE, TB), TB, 0, stream>>>(e2, edst, m2, (long)GE, 1);
    seg_exp_sum_kernel<<<cdiv((long)GE, TB), TB, 0, stream>>>(e2, edst, m2, s2, (long)GE, 1);
    edge_scatter_kernel<<<cdiv((long)GE * 32, TB), TB, 0, stream>>>(e2, s2, xl2, esrc, edst, out2, GE, G_HID, 1);
    bias_add_kernel<<<cdiv((long)GN * G_HID, TB), TB, 0, stream>>>(out2, b2, (long)GN * G_HID, G_HID);

    // ---- action scoring ----
    fill_u64_kernel<<<1, 32, 0, stream>>>(best, 1);
    argmax_col_kernel<<<cdiv(GN, TB), TB, 0, stream>>>(x, best, GN, G_INC, 3);
    score_neighbors_kernel<<<1, 32, 0, stream>>>(best, edst, out2, Ws, bs, out, G_DEG, G_HID);
}
